// TreeLSTMEncoder_21483426414596
// MI455X (gfx1250) — matrix operand order repaired and checked by reading, not compiled, not run on hardware
//
#include <hip/hip_runtime.h>

typedef __attribute__((ext_vector_type(16))) __bf16       v16bf;
typedef __attribute__((ext_vector_type(8)))  float        v8f;
typedef __attribute__((ext_vector_type(4)))  unsigned int u32x4;

#define HDIM 64
#define KDIM 192   // 64 emb + 64 h_left + 64 h_right
#define NOUT 320   // 192 iou + 128 f
#define MTILE 64   // nodes per block (4 subtiles of 16)

union AFrag { v16bf v; u32x4 u[2]; };

// ---- CDNA5 fast transcendentals ------------------------------------------
__device__ __forceinline__ float tanh_fast(float x) {
#if __has_builtin(__builtin_amdgcn_tanhf)
    return __builtin_amdgcn_tanhf(x);      // v_tanh_f32
#else
    return tanhf(x);
#endif
}
__device__ __forceinline__ float sigmoid_fast(float x) {
    return 0.5f * tanh_fast(0.5f * x) + 0.5f;
}

// ---- CDNA5 async global->LDS copy (ASYNCcnt path) ------------------------
__device__ __forceinline__ unsigned lds_off_of(const void* p) {
    return (unsigned)(unsigned long long)(const __attribute__((address_space(3))) char*)p;
}
__device__ __forceinline__ void async_g2l_b128(unsigned ldsOff, const void* g) {
    asm volatile("global_load_async_to_lds_b128 %0, %1, off"
                 :: "v"(ldsOff), "v"(g)
                 : "memory");
}
__device__ __forceinline__ void wait_async0() {
#if __has_builtin(__builtin_amdgcn_s_wait_asynccnt)
    __builtin_amdgcn_s_wait_asynccnt(0);
#else
    asm volatile("s_wait_asynccnt 0x0" ::: "memory");
#endif
}

// ---------------------------------------------------------------------------
// One-shot prep: bf16 embedding table, fused bf16 weight Gt[320][192] laid out
// so a WMMA B-fragment lane reads 16 contiguous K values, fused f32 bias[320].
// ---------------------------------------------------------------------------
__global__ void prep_weights(const float* __restrict__ W_iou, const float* __restrict__ b_Wiou,
                             const float* __restrict__ U_iou, const float* __restrict__ b_Uiou,
                             const float* __restrict__ W_f,   const float* __restrict__ b_Wf,
                             const float* __restrict__ U_f,   const float* __restrict__ b_Uf,
                             const float* __restrict__ emb,   int vocabE,
                             __bf16* __restrict__ emb_bf,
                             __bf16* __restrict__ Gt, float* __restrict__ bias) {
    int t = blockIdx.x * blockDim.x + threadIdx.x;
    if (t < NOUT * KDIM) {
        int nrow = t / KDIM, k = t % KDIM;
        float v;
        if (nrow < 192) {
            v = (k < 64) ? W_iou[nrow * 64 + k] : U_iou[nrow * 128 + (k - 64)];
        } else {
            int fr = nrow - 192;
            v = (k < 64) ? W_f[fr * 64 + k] : U_f[fr * 128 + (k - 64)];
        }
        Gt[t] = (__bf16)v;
    }
    if (t < NOUT)
        bias[t] = (t < 192) ? (b_Wiou[t] + b_Uiou[t]) : (b_Wf[t - 192] + b_Uf[t - 192]);
    if (t < vocabE)
        emb_bf[t] = (__bf16)emb[t];
}

// ---------------------------------------------------------------------------
// One tree level, 64 nodes per block, 20 waves (640 threads).
//  Phase 0: async-stage A = [x | hL | hR] (64 x 192 bf16) into LDS.
//  Phase 1: wave w owns output columns [16w,16w+16); B fragments held in
//           registers; loop 4 M-subtiles x K-steps of v_wmma_f32_16x16x32_bf16
//           with A fragments from LDS (ds_load_b128).
//  Phase 2: per-subtile LSTM cell elementwise (v_tanh_f32 based).
// ---------------------------------------------------------------------------
template <bool LEAF>
__global__ __launch_bounds__(640) void tree_level(
        const int* __restrict__ tok, int start, int nnodes,
        const __bf16* __restrict__ emb_bf,
        const __bf16* __restrict__ Gt,
        const float*  __restrict__ bias,
        const __bf16* __restrict__ h_prev,
        const float*  __restrict__ c_prev,
        __bf16* __restrict__ h_out,
        float*  __restrict__ c_out,
        float*  __restrict__ out_f32) {
    __shared__ __bf16 Astage[MTILE][KDIM];     // 24 KB
    __shared__ float  zbuf[16][NOUT + 8];      // 21 KB, reused per subtile

    const int lane = threadIdx.x & 31;
    const int wave = threadIdx.x >> 5;         // 0..19
    const int base = blockIdx.x * MTILE;

    // ---------------- Phase 0: async staging into LDS ----------------------
    {
        const int t = threadIdx.x;
        if (t < MTILE * 8) {                   // x rows: 64 rows x 8 chunks(16B)
            const int row = t >> 3, ch = t & 7;
            const int nd  = min(base + row, nnodes - 1);
            const __bf16* g = emb_bf + (size_t)tok[start + nd] * HDIM + ch * 8;
            async_g2l_b128(lds_off_of(&Astage[row][ch * 8]), g);
        }
        if (!LEAF) {                           // h rows: 128 contiguous rows x 8 chunks
            for (int t2 = threadIdx.x; t2 < 2 * MTILE * 8; t2 += 640) {
                const int r = t2 >> 3, ch = t2 & 7;
                const int rcl = min(2 * base + r, 2 * nnodes - 1);
                const __bf16* g = h_prev + (size_t)rcl * HDIM + ch * 8;
                async_g2l_b128(lds_off_of(&Astage[r >> 1][64 + (r & 1) * 64 + ch * 8]), g);
            }
        }
        wait_async0();
    }
    __syncthreads();

    // ---------------- Phase 1+2: GEMM subtiles + cell ----------------------
    const int m    = lane & 15;                // A row within subtile
    const int sel  = lane >> 4;                // K-subgroup (contiguous 16-K half)
    const int ncol = wave * 16 + (lane & 15);  // output gate column
    const float bv = bias[ncol];

    constexpr int KSTEPS = LEAF ? 2 : 6;
    // B 32x16 bf16: lane holds 16 consecutive K of its column; keep in regs
    AFrag bfr[KSTEPS];
    {
        const __bf16* bcol = Gt + (size_t)ncol * KDIM + (lane >> 4) * 16;
#pragma unroll
        for (int ks = 0; ks < KSTEPS; ++ks) {
            bfr[ks].u[0] = *(const u32x4*)(bcol + ks * 32);
            bfr[ks].u[1] = *(const u32x4*)(bcol + ks * 32 + 8);
        }
    }

#pragma unroll
    for (int mt = 0; mt < 4; ++mt) {
        v8f acc = {};
        const __bf16* arow = &Astage[mt * 16 + m][0];
#pragma unroll
        for (int ks = 0; ks < KSTEPS; ++ks) {
            AFrag a;   // A 16x32 bf16: lane<16 -> K {kb..kb+15}; lane>=16 -> K {kb+16..kb+31}
            a.u[0] = *(const u32x4*)(arow + ks * 32 + sel * 16);
            a.u[1] = *(const u32x4*)(arow + ks * 32 + sel * 16 + 8);
            acc = __builtin_amdgcn_wmma_f32_16x16x32_bf16(
                    false, a.v, false, bfr[ks].v, (short)0, acc, false, false);
        }

        // C/D tile: lane l, VGPR r -> M = (l>>4)*8 + r, N = l&15
        {
            const int mrow = (lane >> 4) * 8;
#pragma unroll
            for (int r = 0; r < 8; ++r)
                zbuf[mrow + r][ncol] = acc[r] + bv;
        }
        __syncthreads();

        // LSTM cell for this subtile: 16 nodes x 64 hidden
        for (int t = threadIdx.x; t < 16 * HDIM; t += 640) {
            const int mm = t >> 6;
            const int j  = t & (HDIM - 1);
            const int nd = base + mt * 16 + mm;
            if (nd < nnodes) {
                const float ig = sigmoid_fast(zbuf[mm][j]);
                const float og = sigmoid_fast(zbuf[mm][64 + j]);
                const float ug = tanh_fast(sigmoid_fast(zbuf[mm][128 + j])); // ref quirk
                float c = ig * ug;
                if (!LEAF) {
                    const float fl = sigmoid_fast(zbuf[mm][192 + j]);
                    const float fr = sigmoid_fast(zbuf[mm][256 + j]);
                    c += fl * c_prev[(size_t)(2 * nd)     * HDIM + j]
                       + fr * c_prev[(size_t)(2 * nd + 1) * HDIM + j];
                }
                const float h = og * tanh_fast(c);
                h_out[(size_t)nd * HDIM + j] = (__bf16)h;
                c_out[(size_t)nd * HDIM + j] = c;
                if (out_f32) {                 // only level 0 (nnodes==1, nd==0)
                    out_f32[j]        = h;
                    out_f32[HDIM + j] = c;
                }
            }
        }
        __syncthreads();                       // zbuf reuse fence
    }
}

// ---------------------------------------------------------------------------
extern "C" void kernel_launch(void* const* d_in, const int* in_sizes, int n_in,
                              void* d_out, int out_size, void* d_ws, size_t ws_size,
                              hipStream_t stream) {
    const int*   tok    = (const int*)  d_in[0];
    const float* emb    = (const float*)d_in[1];
    const float* W_iou  = (const float*)d_in[2];
    const float* b_Wiou = (const float*)d_in[3];
    const float* U_iou  = (const float*)d_in[4];
    const float* b_Uiou = (const float*)d_in[5];
    const float* W_f    = (const float*)d_in[6];
    const float* b_Wf   = (const float*)d_in[7];
    const float* U_f    = (const float*)d_in[8];
    const float* b_Uf   = (const float*)d_in[9];

    const int nNodes = in_sizes[0];
    int depth = 0;
    while (((1 << depth) - 1) < nNodes) ++depth;   // (1<<depth)-1 == nNodes
    const int vocabE = in_sizes[1];                // VOCAB * E

    // Workspace layout (256B-aligned slices)
    char* w = (char*)d_ws;
    auto take = [&](size_t bytes) -> char* {
        char* p = w;
        w += (bytes + 255) & ~(size_t)255;
        return p;
    };
    __bf16* emb_bf = (__bf16*)take((size_t)vocabE * sizeof(__bf16));
    __bf16* Gt     = (__bf16*)take((size_t)NOUT * KDIM * sizeof(__bf16));
    float*  bias   = (float*) take((size_t)NOUT * sizeof(float));
    const int maxLevel = 1 << (depth - 1);
    __bf16* hbuf[2];
    float*  cbuf[2];
    hbuf[0] = (__bf16*)take((size_t)maxLevel * HDIM * sizeof(__bf16));
    hbuf[1] = (__bf16*)take((size_t)maxLevel * HDIM * sizeof(__bf16));
    cbuf[0] = (float*) take((size_t)maxLevel * HDIM * sizeof(float));
    cbuf[1] = (float*) take((size_t)maxLevel * HDIM * sizeof(float));

    {
        int total  = (vocabE > NOUT * KDIM) ? vocabE : NOUT * KDIM;
        int blocks = (total + 255) / 256;
        prep_weights<<<blocks, 256, 0, stream>>>(W_iou, b_Wiou, U_iou, b_Uiou,
                                                 W_f, b_Wf, U_f, b_Uf,
                                                 emb, vocabE, emb_bf, Gt, bias);
    }

    for (int l = depth - 1; l >= 0; --l) {
        const int start  = (1 << l) - 1;
        const int n      = 1 << l;
        const int blocks = (n + MTILE - 1) / MTILE;
        __bf16* ho   = hbuf[l & 1];
        float*  co   = cbuf[l & 1];
        float*  outp = (l == 0) ? (float*)d_out : nullptr;
        if (l == depth - 1) {
            tree_level<true><<<blocks, 640, 0, stream>>>(
                tok, start, n, emb_bf, Gt, bias,
                nullptr, nullptr, ho, co, outp);
        } else {
            tree_level<false><<<blocks, 640, 0, stream>>>(
                tok, start, n, emb_bf, Gt, bias,
                hbuf[(l + 1) & 1], cbuf[(l + 1) & 1], ho, co, outp);
        }
    }
    (void)n_in; (void)out_size; (void)ws_size;
}
